// ResidualMambaBlock_12747462934751
// MI455X (gfx1250) — matrix operand order repaired and checked by reading, not compiled, not run on hardware
//
#include <hip/hip_runtime.h>
#include <hip/hip_bf16.h>

// ---------------------------------------------------------------------------
// ResidualMambaBlock for MI455X (gfx1250): bf16 WMMA GEMMs + parallel scan.
// B=1, L=2048, D=1024, DI=2048, N=16, DR=64, K=4
// GEMM B-operands pre-swizzled into the CDNA5 WMMA lane layout; each wave
// computes a 32x(16*WT) output block (MT=2 row-tiles reuse every B load).
// ---------------------------------------------------------------------------

#define LSEQ 2048
#define DMODEL 1024
#define DINNER 2048
#define NSTATE 16
#define DRANK 64
#define KCONV 4
#define EPS_RMS 1e-5f

typedef __attribute__((ext_vector_type(16))) __bf16 v16bf;
typedef __attribute__((ext_vector_type(8)))  float  v8f;

// -------------------------------- helpers ----------------------------------

__device__ __forceinline__ float fast_sigmoid(float v) {
    return 1.0f / (1.0f + __expf(-v));
}

__device__ __forceinline__ float softplus_f(float v) {
    return (v > 20.0f) ? v : log1pf(__expf(v));
}

// A tile 16x32 (MxK) bf16 loader; A points at tile origin (row-major, lda elems).
// Contiguous k order (as in the B swizzle): lanes 0-15 row m=lane hold K=0..15; lanes 16-31 hold K=16..31.
__device__ __forceinline__ v16bf load_a16x32(const __bf16* __restrict__ A, int lda, int lane) {
    int row = lane & 15;
    int kh  = (lane >> 4) & 1;
    const __bf16* p = A + row * lda + kh * 16;
    v16bf a;
#pragma unroll
    for (int s = 0; s < 8; ++s) a[s]     = p[s];
#pragma unroll
    for (int s = 0; s < 8; ++s) a[8 + s] = p[8 + s];
    return a;
}

// ------------------------------ WMMA GEMM ----------------------------------
// C[M,N] = act(A[M,K] @ B[K,N] + bias) + resid ; act: 0=none, 1=softplus.
// B swizzled: element (k,n) at dst[((nt*Ktiles + kt)*32 + lane)*16 + s],
//   nt=n>>4, kt=k>>5, lane=((k>>4)&1)*16 + (n&15), s=k&15.
// Wave computes 2 M-tiles x WT N-tiles; every B load feeds 2 WMMAs.
template<int WT>
__global__ void gemm_wmma_bf16(const __bf16* __restrict__ A, int lda,
                               const __bf16* __restrict__ Bsw,
                               float* __restrict__ C, int ldc,
                               const float* __restrict__ bias,
                               const float* __restrict__ resid, int ldr,
                               int M, int N, int K, int act) {
    const int lane    = threadIdx.x & 31;
    const int wave    = threadIdx.x >> 5;
    const int tilesN  = N >> 4;
    const int Ktiles  = K >> 5;
    const int stripsN = tilesN / WT;
    const int total   = (M >> 5) * stripsN;           // 32 rows per strip
    const int strip   = blockIdx.x * (blockDim.x >> 5) + wave;
    if (strip >= total) return;            // wave-uniform: EXEC stays all-ones
    const int tm  = strip / stripsN;                  // 32-row block index
    const int tn0 = (strip - tm * stripsN) * WT;

    // running pointers
    const __bf16* aptr = A + (size_t)(tm << 5) * lda;           // rows tm*32..
    const __bf16* bptr = Bsw + ((size_t)tn0 * Ktiles << 9) + (lane << 4);
    const size_t  wstep = (size_t)Ktiles << 9;                  // per N-tile

    v8f acc[2][WT];
#pragma unroll
    for (int mt = 0; mt < 2; ++mt)
#pragma unroll
        for (int w = 0; w < WT; ++w) acc[mt][w] = (v8f){};

    for (int kt = 0; kt < Ktiles; ++kt) {
        // speculative prefetch of next K-tile (silently dropped past the end)
        __builtin_prefetch(aptr + (lane & 15) * lda + 32, 0, 3);
        __builtin_prefetch(bptr + 512, 0, 3);

        v16bf a0 = load_a16x32(aptr,            lda, lane);
        v16bf a1 = load_a16x32(aptr + 16 * lda, lda, lane);
#pragma unroll
        for (int w = 0; w < WT; ++w) {
            v16bf b = *(const v16bf*)(bptr + (size_t)w * wstep);
            acc[0][w] = __builtin_amdgcn_wmma_f32_16x16x32_bf16(
                false, a0, false, b, (short)0, acc[0][w], false, false);
            acc[1][w] = __builtin_amdgcn_wmma_f32_16x16x32_bf16(
                false, a1, false, b, (short)0, acc[1][w], false, false);
        }
        aptr += 32;        // next K chunk within rows
        bptr += 512;       // next K-tile block (32 lanes * 16 elems)
    }

    // C layout: VGPR i -> row (i + 8*(lane>>4)), col = lane&15
    const int rowh = (lane >> 4) & 1;
#pragma unroll
    for (int w = 0; w < WT; ++w) {
        const int col = ((tn0 + w) << 4) + (lane & 15);
        const float bv = bias ? bias[col] : 0.0f;
#pragma unroll
        for (int mt = 0; mt < 2; ++mt) {
#pragma unroll
            for (int i = 0; i < 8; ++i) {
                const int row = (tm << 5) + (mt << 4) + i + (rowh << 3);
                float v = acc[mt][w][i] + bv;
                if (act == 1) v = softplus_f(v);
                if (resid) v += resid[(size_t)row * ldr + col];
                C[(size_t)row * ldc + col] = v;
            }
        }
    }
}

// ---------------------------- elementwise ----------------------------------

__global__ void cvt_f32_bf16(const float* __restrict__ src, __bf16* __restrict__ dst, int n) {
    int i = blockIdx.x * blockDim.x + threadIdx.x;
    if (i < n) dst[i] = (__bf16)src[i];
}

// Convert f32 (K x N row-major) weight into bf16 WMMA-swizzled B layout.
__global__ void swizzle_b_bf16(const float* __restrict__ src, __bf16* __restrict__ dst,
                               int K, int N) {
    int idx = blockIdx.x * blockDim.x + threadIdx.x;
    if (idx >= K * N) return;
    const int s      = idx & 15;
    const int lane   = (idx >> 4) & 31;
    const int t      = idx >> 9;
    const int Ktiles = K >> 5;
    const int nt = t / Ktiles;
    const int kt = t - nt * Ktiles;
    const int col = (nt << 4) + (lane & 15);
    const int k   = (kt << 5) + ((lane >> 4) << 4) + s;
    dst[idx] = (__bf16)src[(size_t)k * N + col];
}

// RMSNorm: one row (length DMODEL) per block of 256, output bf16
__global__ void rmsnorm_kernel(const float* __restrict__ x, const float* __restrict__ w,
                               __bf16* __restrict__ xn) {
    __shared__ float red[8];
    const int l = blockIdx.x;
    const float* xr = x + (size_t)l * DMODEL;
    float ss = 0.0f;
    for (int i = threadIdx.x; i < DMODEL; i += 256) {
        float v = xr[i];
        ss += v * v;
    }
#pragma unroll
    for (int o = 16; o > 0; o >>= 1) ss += __shfl_xor(ss, o, 32);
    if ((threadIdx.x & 31) == 0) red[threadIdx.x >> 5] = ss;
    __syncthreads();
    float total = 0.0f;
#pragma unroll
    for (int i = 0; i < 8; ++i) total += red[i];
    const float scale = rsqrtf(total / (float)DMODEL + EPS_RMS);
    for (int i = threadIdx.x; i < DMODEL; i += 256) {
        xn[(size_t)l * DMODEL + i] = (__bf16)(xr[i] * scale * w[i]);
    }
}

// Causal depthwise conv(K=4) + bias + SiLU over xs = proj[:, :DINNER]
__global__ void conv_silu_kernel(const float* __restrict__ proj,
                                 const float* __restrict__ conv_w,
                                 const float* __restrict__ conv_b,
                                 float* __restrict__ xc, __bf16* __restrict__ xcb) {
    int idx = blockIdx.x * blockDim.x + threadIdx.x;
    if (idx >= LSEQ * DINNER) return;
    const int c = idx & (DINNER - 1);
    const int l = idx >> 11;
    float acc = conv_b[c];
#pragma unroll
    for (int k = 0; k < KCONV; ++k) {
        int ls = l + k - (KCONV - 1);
        if (ls >= 0) acc += proj[(size_t)ls * (2 * DINNER) + c] * conv_w[c * KCONV + k];
    }
    float s = acc * fast_sigmoid(acc);
    xc[idx]  = s;
    xcb[idx] = (__bf16)s;
}

// Selective scan: thread = (d, n); 256 threads = 16 d x 16 n; grid = DINNER/16.
__global__ void scan_kernel(const float* __restrict__ delta,
                            const float* __restrict__ bcd,
                            const float* __restrict__ xc,
                            const float* __restrict__ log_minus_A,
                            const float* __restrict__ D_param,
                            const float* __restrict__ proj,   // for res slice
                            __bf16* __restrict__ yb) {
    const int n  = threadIdx.x & 15;
    const int dl = threadIdx.x >> 4;
    const int d  = (blockIdx.x << 4) + dl;
    const float Adn = -__expf(log_minus_A[d * NSTATE + n]);
    const float Dp  = D_param[d];
    float h = 0.0f;
    for (int l = 0; l < LSEQ; ++l) {
        const float dlt = delta[(size_t)l * DINNER + d];
        const float xcv = xc[(size_t)l * DINNER + d];
        const float Bm  = bcd[(size_t)l * (2 * NSTATE + DRANK) + DRANK + n];
        const float Cm  = bcd[(size_t)l * (2 * NSTATE + DRANK) + DRANK + NSTATE + n];
        const float abar = __expf(dlt * Adn);
        h = abar * h + dlt * Bm * xcv;
        float c = Cm * h;
        c += __shfl_xor(c, 1, 32);
        c += __shfl_xor(c, 2, 32);
        c += __shfl_xor(c, 4, 32);
        c += __shfl_xor(c, 8, 32);
        if (n == 0) {
            const float r = proj[(size_t)l * (2 * DINNER) + DINNER + d];
            const float y = (c + xcv * Dp) * (r * fast_sigmoid(r));
            yb[(size_t)l * DINNER + d] = (__bf16)y;
        }
    }
}

// ------------------------------- launcher ----------------------------------

extern "C" void kernel_launch(void* const* d_in, const int* in_sizes, int n_in,
                              void* d_out, int out_size, void* d_ws, size_t ws_size,
                              hipStream_t stream) {
    const float* x          = (const float*)d_in[0];   // (L, D)
    const float* rms_w      = (const float*)d_in[1];   // (D,)
    const float* in_proj_w  = (const float*)d_in[2];   // (D, 2*DI)
    const float* conv_w     = (const float*)d_in[3];   // (DI, K)
    const float* conv_b     = (const float*)d_in[4];   // (DI,)
    const float* bcd_w      = (const float*)d_in[5];   // (DI, 2N+DR)
    const float* dup_w      = (const float*)d_in[6];   // (DR, DI)
    const float* dup_b      = (const float*)d_in[7];   // (DI,)
    const float* log_mA     = (const float*)d_in[8];   // (DI, N)
    const float* D_param    = (const float*)d_in[9];   // (DI,)
    const float* out_proj_w = (const float*)d_in[10];  // (DI, D)
    float* out = (float*)d_out;                        // (L, D)

    size_t off = 0;
    auto carve = [&](size_t bytes) -> void* {
        void* p = (char*)d_ws + off;
        off += (bytes + 255) & ~(size_t)255;
        return p;
    };
    __bf16* w_in_sw  = (__bf16*)carve((size_t)DMODEL * 2 * DINNER * 2);
    __bf16* w_bcd_sw = (__bf16*)carve((size_t)DINNER * (2 * NSTATE + DRANK) * 2);
    __bf16* w_dup_sw = (__bf16*)carve((size_t)DRANK * DINNER * 2);
    __bf16* w_out_sw = (__bf16*)carve((size_t)DINNER * DMODEL * 2);
    __bf16* xn_b     = (__bf16*)carve((size_t)LSEQ * DMODEL * 2);
    float*  proj     = (float*) carve((size_t)LSEQ * 2 * DINNER * 4);
    float*  xc       = (float*) carve((size_t)LSEQ * DINNER * 4);
    __bf16* xc_b     = (__bf16*)carve((size_t)LSEQ * DINNER * 2);
    float*  bcd      = (float*) carve((size_t)LSEQ * (2 * NSTATE + DRANK) * 4);
    __bf16* bcd_b    = (__bf16*)carve((size_t)LSEQ * (2 * NSTATE + DRANK) * 2);
    float*  delta    = (float*) carve((size_t)LSEQ * DINNER * 4);
    __bf16* y_b      = (__bf16*)carve((size_t)LSEQ * DINNER * 2);

    // 1) weight conversion + swizzle into WMMA B layout
    auto swz = [&](const float* s, __bf16* d, int K, int N) {
        int n = K * N;
        swizzle_b_bf16<<<(n + 255) / 256, 256, 0, stream>>>(s, d, K, N);
    };
    swz(in_proj_w,  w_in_sw,  DMODEL, 2 * DINNER);
    swz(bcd_w,      w_bcd_sw, DINNER, 2 * NSTATE + DRANK);
    swz(dup_w,      w_dup_sw, DRANK,  DINNER);
    swz(out_proj_w, w_out_sw, DINNER, DMODEL);

    // 2) RMSNorm -> xn (bf16)
    rmsnorm_kernel<<<LSEQ, 256, 0, stream>>>(x, rms_w, xn_b);

    const int WPB = 8;  // waves per block (256 threads)
    auto gemm = [&](const __bf16* A, int lda, const __bf16* Bsw,
                    float* C, int ldc, const float* bias,
                    const float* resid, int ldr, int M, int N, int K, int act) {
        int tilesN = N >> 4;
        if ((tilesN & 3) == 0) {
            int strips = (M >> 5) * (tilesN >> 2);
            gemm_wmma_bf16<4><<<(strips + WPB - 1) / WPB, WPB * 32, 0, stream>>>(
                A, lda, Bsw, C, ldc, bias, resid, ldr, M, N, K, act);
        } else {
            int strips = (M >> 5) * (tilesN >> 1);
            gemm_wmma_bf16<2><<<(strips + WPB - 1) / WPB, WPB * 32, 0, stream>>>(
                A, lda, Bsw, C, ldc, bias, resid, ldr, M, N, K, act);
        }
    };

    // 3) in_proj GEMM: (L,D) @ (D,2DI) -> proj
    gemm(xn_b, DMODEL, w_in_sw, proj, 2 * DINNER,
         nullptr, nullptr, 0, LSEQ, 2 * DINNER, DMODEL, 0);

    // 4) causal conv + SiLU -> xc (f32 + bf16)
    conv_silu_kernel<<<(LSEQ * DINNER + 255) / 256, 256, 0, stream>>>(
        proj, conv_w, conv_b, xc, xc_b);

    // 5) bcd GEMM: (L,DI) @ (DI,96) -> bcd ; bf16 copy feeds delta GEMM's A
    gemm(xc_b, DINNER, w_bcd_sw, bcd, 2 * NSTATE + DRANK,
         nullptr, nullptr, 0, LSEQ, 2 * NSTATE + DRANK, DINNER, 0);
    {
        int n = LSEQ * (2 * NSTATE + DRANK);
        cvt_f32_bf16<<<(n + 255) / 256, 256, 0, stream>>>(bcd, bcd_b, n);
    }

    // 6) delta GEMM: delta_lr(L,64 stride 96) @ (64,DI) + dup_b, softplus
    gemm(bcd_b, 2 * NSTATE + DRANK, w_dup_sw, delta, DINNER,
         dup_b, nullptr, 0, LSEQ, DINNER, DRANK, 1);

    // 7) selective scan + gating -> y (bf16)
    scan_kernel<<<DINNER / 16, 256, 0, stream>>>(delta, bcd, xc, log_mA,
                                                 D_param, proj, y_b);

    // 8) out_proj GEMM + residual x -> out
    gemm(y_b, DINNER, w_out_sw, out, DMODEL,
         nullptr, x, DMODEL, LSEQ, DMODEL, DINNER, 0);

    (void)in_sizes; (void)n_in; (void)out_size; (void)ws_size;
}
